// ResNet_36876589203549
// MI455X (gfx1250) — matrix operand order repaired and checked by reading, not compiled, not run on hardware
//
#include <hip/hip_runtime.h>

// Problem constants (match reference)
#define BATCH 32
#define CIN   64
#define COUT  128
#define IN_H  112
#define IN_W  112
#define PW    114            // padded H/W
#define KCHUNK 32            // bf16 WMMA K
#define NCHUNK 18            // 576 / 32 K-chunks
#define NTILE 112            // spatial columns per output row
#define NROWS 2              // output rows per block -> N = 224
#define LDSTRIDE 40          // padded LDS row stride in bf16 elems (80B, conflict-free)

typedef __attribute__((ext_vector_type(16))) __bf16 v16bf;
typedef __attribute__((ext_vector_type(8)))  __bf16 v8bf;
typedef __attribute__((ext_vector_type(8)))  float  v8f;

__device__ __forceinline__ unsigned short f2bf_bits(float f) {
    union { float f; unsigned u; } v; v.f = f;
    return (unsigned short)((v.u + 0x7FFFu + ((v.u >> 16) & 1u)) >> 16);
}

__device__ __forceinline__ unsigned pack_bf16x2(float lo, float hi) {
#if __has_builtin(__builtin_amdgcn_cvt_pk_bf16_f32)
    typedef __attribute__((ext_vector_type(2))) __bf16 v2bf;
    union { v2bf v; unsigned u; } r;
    r.v = __builtin_amdgcn_cvt_pk_bf16_f32(lo, hi);
    return r.u;
#else
    return (unsigned)f2bf_bits(lo) | ((unsigned)f2bf_bits(hi) << 16);
#endif
}

// ---------------------------------------------------------------------------
// One-shot weight transform: f32 [COUT][CIN][3][3] -> bf16 per-chunk LDS image
//   wt[chunk][m][kkpad]  with chunk = (kh*3+kw)*2 + half, K = r*64 + half*32+kk
// ---------------------------------------------------------------------------
__global__ __launch_bounds__(256)
void wt_transform(const float* __restrict__ w, unsigned short* __restrict__ wt) {
    const int total = NCHUNK * COUT * LDSTRIDE;           // 92160
    int idx = blockIdx.x * 256 + threadIdx.x;
    if (idx >= total) return;
    int chunk = idx / (COUT * LDSTRIDE);
    int rem   = idx - chunk * (COUT * LDSTRIDE);
    int m     = rem / LDSTRIDE;
    int kk    = rem - m * LDSTRIDE;
    if (kk >= KCHUNK) { wt[idx] = 0; return; }
    int r   = chunk >> 1;                                  // kh*3 + kw
    int cin = ((chunk & 1) << 5) + kk;
    wt[idx] = f2bf_bits(w[m * (CIN * 9) + cin * 9 + r]);
}

// ---------------------------------------------------------------------------
// Implicit-GEMM conv: block = one (b, h-pair): 128 x 224 GEMM tile
// ---------------------------------------------------------------------------
__global__ __launch_bounds__(256)
void conv3x3_wmma_bf16(const float* __restrict__ x,            // [B,CIN,114,114]
                       const unsigned short* __restrict__ wt,  // bf16 chunk images
                       const float* __restrict__ bias,         // [COUT]
                       float* __restrict__ out)                // [B,COUT,112,112]
{
    __shared__ __align__(64) unsigned short w_lds[COUT * LDSTRIDE];          // 10240 B
    __shared__ __align__(64) unsigned short x_lds[NROWS * NTILE * LDSTRIDE]; // 17920 B

    const int tid  = threadIdx.x;
    const int wave = tid >> 5;
    const int lane = tid & 31;
    const int lrow = lane & 15;
    const int kgrp = lane >> 4;          // K-half per ISA A/B fragment layouts

    const int b  = blockIdx.x / (IN_H / NROWS);
    const int h0 = (blockIdx.x % (IN_H / NROWS)) * NROWS;
    const int mbase = wave * 16;

    v8f acc[7 * NROWS];
#pragma unroll
    for (int i = 0; i < 7 * NROWS; ++i)
        acc[i] = (v8f){0.f,0.f,0.f,0.f,0.f,0.f,0.f,0.f};

    const float* xb = x + (size_t)b * CIN * PW * PW;

    // input staging coords: thread owns cin-pair kk2 and 16-strided columns
    const int kk2 = (tid >> 4) << 1;     // even kk in 0..30
    const int jb  = tid & 15;            // column phase 0..15

    // A fragment source: row M = mbase+lrow; element offsets {8g..8g+7, 8g+16..}
    const unsigned short* aptr = &w_lds[(mbase + lrow) * LDSTRIDE + 8 * kgrp];
    // B fragment source: K-half select
    const unsigned short* bbase = &x_lds[8 * kgrp];

#pragma unroll
    for (int r = 0; r < 9; ++r) {        // r = kh*3 + kw (compile-time)
        const int kh = r / 3;
        const int kw = r - kh * 3;
#pragma unroll
        for (int half = 0; half < 2; ++half) {
            const int chunk = r * 2 + half;

            // ---- stage weights: straight 10240B copy (no math, coalesced) ----
            const uint4* wsrc = (const uint4*)(wt + (size_t)chunk * COUT * LDSTRIDE);
            uint4* wdst = (uint4*)w_lds;
            wdst[tid]       = wsrc[tid];
            wdst[tid + 256] = wsrc[tid + 256];
            if (tid < 128) wdst[tid + 512] = wsrc[tid + 512];

            // ---- stage input: 2 rows x 32 cin x 112 cols, packed bf16x2 ----
#pragma unroll
            for (int rh = 0; rh < NROWS; ++rh) {
                const float* xrow = xb + (size_t)(h0 + rh + kh) * PW + kw;
                const float* src0 = xrow + (size_t)((half << 5) + kk2)     * (PW * PW);
                const float* src1 = xrow + (size_t)((half << 5) + kk2 + 1) * (PW * PW);
                unsigned* xdst = (unsigned*)&x_lds[rh * NTILE * LDSTRIDE + kk2];
#pragma unroll
                for (int i = 0; i < 7; ++i) {
                    int j = jb + i * 16;               // 16 lanes cover 112 cols
                    xdst[(j * LDSTRIDE) >> 1] = pack_bf16x2(src0[j], src1[j]);
                }
            }

            if (chunk + 1 < NCHUNK)  // pull next weight image toward L2/WGP$
                __builtin_prefetch(wt + (size_t)(chunk + 1) * COUT * LDSTRIDE + tid * 16, 0, 1);

            __syncthreads();

            // ---- A fragment: two 16B LDS loads + shuffle (reused by 14 WMMAs) ----
            v8bf alo = *(const v8bf*)(aptr);
            v8bf ahi = *(const v8bf*)(aptr + 16);
            v16bf afrag = __builtin_shufflevector(alo, ahi,
                            0,1,2,3,4,5,6,7,8,9,10,11,12,13,14,15);

            // ---- 14 N-subtiles across the two output rows ----
#pragma unroll
            for (int ns = 0; ns < 7 * NROWS; ++ns) {
                const unsigned short* bp = bbase + (ns * 16 + lrow) * LDSTRIDE;
                v8bf blo = *(const v8bf*)(bp);
                v8bf bhi = *(const v8bf*)(bp + 16);
                v16bf bfrag = __builtin_shufflevector(blo, bhi,
                                0,1,2,3,4,5,6,7,8,9,10,11,12,13,14,15);
                acc[ns] = __builtin_amdgcn_wmma_f32_16x16x32_bf16(
                              false, afrag, false, bfrag,
                              (short)0, acc[ns], false, false);
            }
            __syncthreads();
        }
    }

    // ---- epilogue: C/D layout VGPR r -> M = r + 8*kgrp, N = lane&15 ----
#pragma unroll
    for (int rr = 0; rr < 8; ++rr) {
        int m = mbase + rr + 8 * kgrp;
        float bv = bias[m];
#pragma unroll
        for (int rh = 0; rh < NROWS; ++rh) {
            float* orow = out + (((size_t)b * COUT + m) * IN_H + (h0 + rh)) * IN_W;
#pragma unroll
            for (int ns = 0; ns < 7; ++ns) {
                orow[ns * 16 + lrow] = acc[rh * 7 + ns][rr] + bv;
            }
        }
    }
}

extern "C" void kernel_launch(void* const* d_in, const int* in_sizes, int n_in,
                              void* d_out, int out_size, void* d_ws, size_t ws_size,
                              hipStream_t stream) {
    // setup_inputs order: x_padded, in_height, in_width, weight, bias
    const float* x    = (const float*)d_in[0];
    const float* wgt  = (const float*)d_in[3];
    const float* bias = (const float*)d_in[4];
    float* out = (float*)d_out;

    // d_ws: 18*128*40 bf16 = 184320 B weight chunk images
    unsigned short* wt = (unsigned short*)d_ws;

    {   // one-shot weight transform (deterministic, re-runs every call)
        int total  = NCHUNK * COUT * LDSTRIDE;
        int blocks = (total + 255) / 256;
        wt_transform<<<blocks, 256, 0, stream>>>(wgt, wt);
    }

    dim3 grid(BATCH * (IN_H / NROWS));  // 1792 blocks: one (b, h-pair) each
    dim3 block(256);                    // 8 wave32 waves
    conv3x3_wmma_bf16<<<grid, block, 0, stream>>>(x, wt, bias, out);
}
